// T5MoEBlock_82695300317839
// MI455X (gfx1250) — matrix operand order repaired and checked by reading, not compiled, not run on hardware
//
#include <hip/hip_runtime.h>
#include <cstdint>
#include <cstddef>

typedef __bf16 bf16;
typedef __attribute__((ext_vector_type(16))) __bf16 v16bf;
typedef __attribute__((ext_vector_type(8)))  __bf16 v8bf;
typedef __attribute__((ext_vector_type(4)))  __bf16 v4bf;
typedef __attribute__((ext_vector_type(8)))  float  v8f;

#define T_TOK   8192
#define DMODEL  768
#define FDIM    3072
#define NEXP    8
#define PADCAP  16640            /* 2*T + NEXP*32 padding */
#define ROWTILES (PADCAP / 32)   /* 520 */
#define KC      64               /* K-chunk per stage (2 WMMA k-steps) */
#define KSTR    72               /* LDS row stride (bf16 elems), 144B: 16B-aligned, bank-spread */

__device__ inline v16bf cat8(v8bf lo, v8bf hi) {
    return __builtin_shufflevector(lo, hi, 0,1,2,3,4,5,6,7,8,9,10,11,12,13,14,15);
}

// ---------------------------------------------------------------------------
// Kernel 1: router logits, softmax-top2 renorm, RMS rstd, expert counting.
// One wave (32 lanes) per token; 8 tokens per 256-thread block.
// ---------------------------------------------------------------------------
__global__ __launch_bounds__(256)
void router_kernel(const float* __restrict__ x, const float* __restrict__ rw,
                   float* __restrict__ logits, int* __restrict__ cnt,
                   int2* __restrict__ sel, float2* __restrict__ wgt,
                   float* __restrict__ rstd)
{
    __shared__ float RW[NEXP * DMODEL];  // 24 KB
    const int tid = threadIdx.x;
    for (int i = tid; i < NEXP * DMODEL; i += 256) RW[i] = rw[i];
    __syncthreads();

    const int wave = tid >> 5, lane = tid & 31;
    const int t = blockIdx.x * 8 + wave;
    const float* xt = x + (size_t)t * DMODEL;

    float acc[NEXP];
    #pragma unroll
    for (int e = 0; e < NEXP; ++e) acc[e] = 0.f;
    float ssq = 0.f;

    #pragma unroll 4
    for (int i = 0; i < DMODEL / 32; ++i) {
        const int d = lane + i * 32;
        const float v = xt[d];
        ssq += v * v;
        #pragma unroll
        for (int e = 0; e < NEXP; ++e) acc[e] += v * RW[e * DMODEL + d];
    }
    #pragma unroll
    for (int off = 16; off > 0; off >>= 1) {
        ssq += __shfl_xor(ssq, off, 32);
        #pragma unroll
        for (int e = 0; e < NEXP; ++e) acc[e] += __shfl_xor(acc[e], off, 32);
    }

    if (lane == 0) {
        #pragma unroll
        for (int e = 0; e < NEXP; ++e) logits[(size_t)t * NEXP + e] = acc[e];
        rstd[t] = rsqrtf(ssq * (1.f / DMODEL) + 1e-6f);
        int e0 = 0;
        #pragma unroll
        for (int e = 1; e < NEXP; ++e) if (acc[e] > acc[e0]) e0 = e;
        int e1 = (e0 == 0) ? 1 : 0;
        #pragma unroll
        for (int e = 0; e < NEXP; ++e)
            if (e != e0 && acc[e] > acc[e1]) e1 = e;
        const float p1 = __expf(acc[e1] - acc[e0]);  // softmax denom cancels in top-2 renorm
        const float w0 = 1.f / (1.f + p1);
        sel[t] = make_int2(e0, e1);
        wgt[t] = make_float2(w0, 1.f - w0);
        atomicAdd(&cnt[e0], 1);
        atomicAdd(&cnt[e1], 1);
    }
}

// ---------------------------------------------------------------------------
// Kernel 2: precompute bf16 x_norm = x * rsqrt(mean(x^2)+eps). One float4 per
// thread -> global_load_b128 + ds-free v_cvt_pk_bf16 + 8B store.
// ---------------------------------------------------------------------------
__global__ __launch_bounds__(256)
void xnorm_kernel(const float* __restrict__ x, const float* __restrict__ rstd,
                  bf16* __restrict__ xn)
{
    const int idx = blockIdx.x * 256 + threadIdx.x;   // float4 granularity
    const int t = idx / (DMODEL / 4);
    const int d4 = idx - t * (DMODEL / 4);
    const float r = rstd[t];
    const float4 v = ((const float4*)x)[(size_t)t * (DMODEL / 4) + d4];
    v4bf o;
    o[0] = (bf16)(v.x * r); o[1] = (bf16)(v.y * r);
    o[2] = (bf16)(v.z * r); o[3] = (bf16)(v.w * r);
    *(v4bf*)(xn + (size_t)t * DMODEL + d4 * 4) = o;
}

// ---------------------------------------------------------------------------
// Kernel 3: exclusive prefix scan over 8 expert counts, padded to 32 rows.
// ---------------------------------------------------------------------------
__global__ void scan_kernel(const int* __restrict__ cnt, int* __restrict__ cursor,
                            int* __restrict__ offs)
{
    int o = 0;
    for (int e = 0; e < NEXP; ++e) {
        offs[e] = o;
        cursor[e] = o;
        o += (cnt[e] + 31) & ~31;
    }
    offs[NEXP] = o;
}

// ---------------------------------------------------------------------------
// Kernel 4: build compact per-expert token lists. list entry = (token<<1)|slot.
// ---------------------------------------------------------------------------
__global__ __launch_bounds__(256)
void fill_kernel(const int2* __restrict__ sel, int* __restrict__ cursor,
                 int* __restrict__ list)
{
    const int t = blockIdx.x * 256 + threadIdx.x;
    const int2 s = sel[t];
    const int p0 = atomicAdd(&cursor[s.x], 1);
    list[p0] = (t << 1);
    const int p1 = atomicAdd(&cursor[s.y], 1);
    list[p1] = (t << 1) | 1;
}

// ---------------------------------------------------------------------------
// Kernel 5: GEMM1  H = relu( x_norm @ (wi[e]*ln_w[e]).T )  -> hbuf (bf16)
// Block: 32 gathered tokens x 128 F-cols, K=768 in KC=64 double-buffered
// stages. 8 waves: each owns one 16-row half and two 16-col WMMA tiles.
// ---------------------------------------------------------------------------
__global__ __launch_bounds__(256)
void gemm1_kernel(const bf16* __restrict__ xn, const float* __restrict__ wi,
                  const float* __restrict__ lnw, const int* __restrict__ offs,
                  const int* __restrict__ cnt, const int* __restrict__ list,
                  bf16* __restrict__ hbuf)
{
    __shared__ __align__(16) bf16 Xs[2][32][KSTR];    //  9.2 KB
    __shared__ __align__(16) bf16 Bs[2][128][KSTR];   // 36.9 KB
    __shared__ int sTok[32];

    const int gr0 = blockIdx.x * 32;
    const int total = offs[NEXP];
    if (gr0 >= total) return;
    int e = 0;
    while (offs[e + 1] <= gr0) ++e;
    const int lim = offs[e] + cnt[e];
    const int f0 = blockIdx.y * 128;
    const int tid = threadIdx.x;

    if (tid < 32) {
        const int gr = gr0 + tid;
        sTok[tid] = (gr < lim) ? (list[gr] >> 1) : -1;
    }
    __syncthreads();

    const size_t wrow = ((size_t)e * FDIM + f0) * DMODEL;   // wi[e][f0][0]
    const float* lne = lnw + (size_t)e * DMODEL;

    // ---- staging helper (fully unrolled, vector loads/stores) ----
    auto stage = [&](int kcs, int buf) {
        // A tile: 32 rows x 64 bf16 = one uint4 (8 elems) per thread
        {
            const int r = tid >> 3, c = (tid & 7) * 8;
            const int t = sTok[r];
            uint4 v = make_uint4(0u, 0u, 0u, 0u);
            if (t >= 0)
                v = *(const uint4*)(xn + (size_t)t * DMODEL + kcs * KC + c);
            *(uint4*)&Xs[buf][r][c] = v;
        }
        // B tile: 128 rows x 64 f32 -> bf16 (ln_w folded); 8 float4 per thread
        #pragma unroll
        for (int p = 0; p < 8; ++p) {
            const int idx = p * 256 + tid;           // 0..2047 float4s
            const int n = idx >> 4, j = (idx & 15) * 4;
            const float4 wv = *(const float4*)&wi[wrow + (size_t)n * DMODEL + kcs * KC + j];
            const float4 lv = *(const float4*)&lne[kcs * KC + j];
            v4bf o;
            o[0] = (bf16)(wv.x * lv.x); o[1] = (bf16)(wv.y * lv.y);
            o[2] = (bf16)(wv.z * lv.z); o[3] = (bf16)(wv.w * lv.w);
            *(v4bf*)&Bs[buf][n][j] = o;
        }
    };

    const int wave = tid >> 5, lane = tid & 31;
    const int half = lane >> 4, l15 = lane & 15;
    const int mtile = wave >> 2;            // 0..1
    const int nb    = (wave & 3) * 2;       // 0,2,4,6

    v8f acc0 = {0.f,0.f,0.f,0.f,0.f,0.f,0.f,0.f};
    v8f acc1 = {0.f,0.f,0.f,0.f,0.f,0.f,0.f,0.f};

    stage(0, 0);
    __syncthreads();

    const int NIT = DMODEL / KC;            // 12
    for (int kc = 0; kc < NIT; ++kc) {
        if (kc + 1 < NIT) stage(kc + 1, (kc + 1) & 1);
        const int cb = kc & 1;
        #pragma unroll
        for (int ks = 0; ks < 2; ++ks) {
            const bf16* xr = &Xs[cb][mtile * 16 + l15][ks * 32 + half * 16];
            const v16bf a = cat8(*(const v8bf*)xr, *(const v8bf*)(xr + 8));
            const bf16* b0 = &Bs[cb][nb * 16 + l15][ks * 32 + half * 16];
            acc0 = __builtin_amdgcn_wmma_f32_16x16x32_bf16(
                false, a, false, cat8(*(const v8bf*)b0, *(const v8bf*)(b0 + 8)),
                (short)0, acc0, false, false);
            const bf16* b1 = &Bs[cb][(nb + 1) * 16 + l15][ks * 32 + half * 16];
            acc1 = __builtin_amdgcn_wmma_f32_16x16x32_bf16(
                false, a, false, cat8(*(const v8bf*)b1, *(const v8bf*)(b1 + 8)),
                (short)0, acc1, false, false);
        }
        __syncthreads();
    }

    // Epilogue: ReLU + bf16 store into hbuf (padded row space, always in-bounds).
    #pragma unroll
    for (int i = 0; i < 8; ++i) {
        const int m = mtile * 16 + half * 8 + i;
        const size_t row = (size_t)(gr0 + m) * FDIM + f0;
        hbuf[row + nb * 16 + l15]       = (bf16)fmaxf(acc0[i], 0.f);
        hbuf[row + (nb + 1) * 16 + l15] = (bf16)fmaxf(acc1[i], 0.f);
    }
}

// ---------------------------------------------------------------------------
// Kernel 6: GEMM2  FF = H @ wo[e].T  -> ffbuf (f32, per (token,slot) row)
// Block: 32 rows x 128 D-cols, K=3072 in KC=64 double-buffered stages.
// ---------------------------------------------------------------------------
__global__ __launch_bounds__(256)
void gemm2_kernel(const bf16* __restrict__ hbuf, const float* __restrict__ wo,
                  const int* __restrict__ offs, const int* __restrict__ cnt,
                  const int* __restrict__ list, float* __restrict__ ffbuf)
{
    __shared__ __align__(16) bf16 Hs[2][32][KSTR];
    __shared__ __align__(16) bf16 Ws[2][128][KSTR];
    __shared__ int sPk[32];

    const int gr0 = blockIdx.x * 32;
    const int total = offs[NEXP];
    if (gr0 >= total) return;
    int e = 0;
    while (offs[e + 1] <= gr0) ++e;
    const int lim = offs[e] + cnt[e];
    const int d0 = blockIdx.y * 128;
    const int tid = threadIdx.x;

    if (tid < 32) {
        const int gr = gr0 + tid;
        sPk[tid] = (gr < lim) ? list[gr] : -1;
    }
    __syncthreads();

    const size_t wrow = ((size_t)e * DMODEL + d0) * FDIM;   // wo[e][d0][0]

    auto stage = [&](int kcs, int buf) {
        // A tile: H rows, raw bf16 copy; one uint4 per thread
        {
            const int r = tid >> 3, c = (tid & 7) * 8;
            *(uint4*)&Hs[buf][r][c] =
                *(const uint4*)(hbuf + (size_t)(gr0 + r) * FDIM + kcs * KC + c);
        }
        // B tile: 128 rows x 64 f32 -> bf16; 8 float4 per thread
        #pragma unroll
        for (int p = 0; p < 8; ++p) {
            const int idx = p * 256 + tid;
            const int n = idx >> 4, j = (idx & 15) * 4;
            const float4 wv = *(const float4*)&wo[wrow + (size_t)n * FDIM + kcs * KC + j];
            v4bf o;
            o[0] = (bf16)wv.x; o[1] = (bf16)wv.y;
            o[2] = (bf16)wv.z; o[3] = (bf16)wv.w;
            *(v4bf*)&Ws[buf][n][j] = o;
        }
    };

    const int wave = tid >> 5, lane = tid & 31;
    const int half = lane >> 4, l15 = lane & 15;
    const int mtile = wave >> 2;
    const int nb    = (wave & 3) * 2;

    v8f acc0 = {0.f,0.f,0.f,0.f,0.f,0.f,0.f,0.f};
    v8f acc1 = {0.f,0.f,0.f,0.f,0.f,0.f,0.f,0.f};

    stage(0, 0);
    __syncthreads();

    const int NIT = FDIM / KC;              // 48
    for (int kc = 0; kc < NIT; ++kc) {
        if (kc + 1 < NIT) stage(kc + 1, (kc + 1) & 1);
        const int cb = kc & 1;
        #pragma unroll
        for (int ks = 0; ks < 2; ++ks) {
            const bf16* xr = &Hs[cb][mtile * 16 + l15][ks * 32 + half * 16];
            const v16bf a = cat8(*(const v8bf*)xr, *(const v8bf*)(xr + 8));
            const bf16* b0 = &Ws[cb][nb * 16 + l15][ks * 32 + half * 16];
            acc0 = __builtin_amdgcn_wmma_f32_16x16x32_bf16(
                false, a, false, cat8(*(const v8bf*)b0, *(const v8bf*)(b0 + 8)),
                (short)0, acc0, false, false);
            const bf16* b1 = &Ws[cb][(nb + 1) * 16 + l15][ks * 32 + half * 16];
            acc1 = __builtin_amdgcn_wmma_f32_16x16x32_bf16(
                false, a, false, cat8(*(const v8bf*)b1, *(const v8bf*)(b1 + 8)),
                (short)0, acc1, false, false);
        }
        __syncthreads();
    }

    // Scatter FF rows keyed by (token<<1)|slot -> deterministic placement.
    #pragma unroll
    for (int i = 0; i < 8; ++i) {
        const int m = mtile * 16 + half * 8 + i;
        const int pk = sPk[m];
        if (pk >= 0) {
            const size_t row = (size_t)pk * DMODEL + d0;
            ffbuf[row + nb * 16 + l15]       = acc0[i];
            ffbuf[row + (nb + 1) * 16 + l15] = acc1[i];
        }
    }
}

// ---------------------------------------------------------------------------
// Kernel 7: combine  y = w0*(x+ff0) + w1*(x+ff1)   (float4 vectorized)
// ---------------------------------------------------------------------------
__global__ __launch_bounds__(256)
void combine_kernel(const float* __restrict__ x, const float* __restrict__ ff,
                    const float2* __restrict__ wgt, float* __restrict__ out)
{
    const int idx = blockIdx.x * 256 + threadIdx.x;     // one float4 per thread
    const int t = idx / (DMODEL / 4);
    const int d4 = idx - t * (DMODEL / 4);
    const float4 xv = ((const float4*)x)[(size_t)t * (DMODEL / 4) + d4];
    const float4 a  = ((const float4*)ff)[((size_t)t * 2)     * (DMODEL / 4) + d4];
    const float4 b  = ((const float4*)ff)[((size_t)t * 2 + 1) * (DMODEL / 4) + d4];
    const float2 w = wgt[t];
    float4 o;
    o.x = w.x * (xv.x + a.x) + w.y * (xv.x + b.x);
    o.y = w.x * (xv.y + a.y) + w.y * (xv.y + b.y);
    o.z = w.x * (xv.z + a.z) + w.y * (xv.z + b.z);
    o.w = w.x * (xv.w + a.w) + w.y * (xv.w + b.w);
    ((float4*)out)[(size_t)t * (DMODEL / 4) + d4] = o;
}

// ---------------------------------------------------------------------------
extern "C" void kernel_launch(void* const* d_in, const int* in_sizes, int n_in,
                              void* d_out, int out_size, void* d_ws, size_t ws_size,
                              hipStream_t stream)
{
    (void)in_sizes; (void)n_in; (void)out_size; (void)ws_size;
    const float* x   = (const float*)d_in[0];   // [T, D]
    const float* rw  = (const float*)d_in[1];   // [E, D]
    const float* lnw = (const float*)d_in[2];   // [E, D]
    const float* wi  = (const float*)d_in[3];   // [E, F, D]
    const float* wo  = (const float*)d_in[4];   // [E, D, F]

    float* out    = (float*)d_out;                              // [T, D]
    float* logits = out + (size_t)T_TOK * DMODEL;               // [T, E]

    // Workspace carve-out (256B aligned regions).
    char* p = (char*)d_ws;
    auto carve = [&](size_t bytes) -> void* {
        void* r = (void*)p;
        p += (bytes + 255) & ~(size_t)255;
        return r;
    };
    int*    cnt    = (int*)carve(NEXP * sizeof(int));
    int*    cursor = (int*)carve(NEXP * sizeof(int));
    int*    offs   = (int*)carve((NEXP + 1) * sizeof(int));
    int2*   sel    = (int2*)carve((size_t)T_TOK * sizeof(int2));
    float2* wgt    = (float2*)carve((size_t)T_TOK * sizeof(float2));
    float*  rstd   = (float*)carve((size_t)T_TOK * sizeof(float));
    int*    list   = (int*)carve((size_t)PADCAP * sizeof(int));
    bf16*   xnbuf  = (bf16*)carve((size_t)T_TOK * DMODEL * sizeof(bf16));
    bf16*   hbuf   = (bf16*)carve((size_t)PADCAP * FDIM * sizeof(bf16));
    float*  ffbuf  = (float*)carve((size_t)2 * T_TOK * DMODEL * sizeof(float));

    // Zero the counters/offsets (graph-capture safe).
    hipMemsetAsync(cnt, 0, 3 * 256, stream);

    router_kernel<<<T_TOK / 8, 256, 0, stream>>>(x, rw, logits, cnt, sel, wgt, rstd);
    xnorm_kernel<<<(T_TOK * DMODEL / 4) / 256, 256, 0, stream>>>(x, rstd, xnbuf);
    scan_kernel<<<1, 1, 0, stream>>>(cnt, cursor, offs);
    fill_kernel<<<T_TOK / 256, 256, 0, stream>>>(sel, cursor, list);
    gemm1_kernel<<<dim3(ROWTILES, FDIM / 128), 256, 0, stream>>>(
        xnbuf, wi, lnw, offs, cnt, list, hbuf);
    gemm2_kernel<<<dim3(ROWTILES, DMODEL / 128), 256, 0, stream>>>(
        hbuf, wo, offs, cnt, list, ffbuf);
    combine_kernel<<<(T_TOK * DMODEL / 4) / 256, 256, 0, stream>>>(x, ffbuf, wgt, out);
}
